// MultiHeadAttention_49143015801007
// MI455X (gfx1250) — matrix operand order repaired and checked by reading, not compiled, not run on hardware
//
#include <hip/hip_runtime.h>
#include <hip/hip_bf16.h>
#include <cstdint>
#include <cmath>

// ---------------------------------------------------------------------------
// MI455X (gfx1250, wave32): embed -> biLSTM -> MHA -> FC.
// All GEMMs on v_wmma_f32_16x16x32_f16. Operand layouts are chosen so every
// WMMA fragment is loaded as contiguous 16B chunks (no per-half LDS gathers).
// Global->LDS staging uses gfx1250 async loads (ASYNCcnt).
// ---------------------------------------------------------------------------

typedef __attribute__((ext_vector_type(16))) _Float16 v16h;
typedef __attribute__((ext_vector_type(8)))  float    v8f;
typedef _Float16 h16;

static constexpr int B_  = 64;
static constexpr int S_  = 512;
static constexpr int E_  = 128;
static constexpr int H_  = 64;
static constexpr int G_  = 256;    // 4*H
static constexpr int NH_ = 8;
static constexpr int TAR_ = 14;
static constexpr int T_  = B_ * S_;     // 32768 tokens
static constexpr int D2_ = 2 * H_;      // 128
static constexpr int QN_ = NH_ * 64;    // 512 (all heads)

union Frag16 { uint4 u[2]; v16h v; };     // 16 halves
union Frag8h { h16 h[8]; uint4 u; };      // 8 halves
union Frag8f { uint4 u[2]; v8f f; };      // 8 floats

__device__ __forceinline__ v8f wmma_f16(v16h a, v16h b, v8f c) {
  return __builtin_amdgcn_wmma_f32_16x16x32_f16(false, a, false, b, (short)0, c,
                                                false, false);
}

__device__ __forceinline__ float sigf(float x) { return 1.f / (1.f + __expf(-x)); }

// Per-lane async copy of 16 bytes global -> LDS (GLOBAL_LOAD_ASYNC_TO_LDS_B128)
__device__ __forceinline__ void async_b128(void* lds, const void* gmem) {
  uint32_t la = (uint32_t)(uintptr_t)lds;        // LDS byte address
  uint64_t ga = (uint64_t)(uintptr_t)gmem;       // global address
  asm volatile("global_load_async_to_lds_b128 %0, %1, off"
               :: "v"(la), "v"(ga) : "memory");
}
__device__ __forceinline__ void async_wait() {
  asm volatile("s_wait_asynccnt 0x0" ::: "memory");
}

// ---------------------------------------------------------------------------
// Embedding gather (f32 table -> f16 activations)
// ---------------------------------------------------------------------------
__global__ void k_embed_gather(const int* __restrict__ seq,
                               const float* __restrict__ emb,
                               h16* __restrict__ xh) {
  int tok = blockIdx.x;
  int e = threadIdx.x;  // 128
  int v = seq[tok];
  xh[(size_t)tok * E_ + e] = (h16)emb[(size_t)v * E_ + e];
}

// straight f32 -> f16 (layout preserved: weights stay [N,K])
__global__ void k_tof16(const float* __restrict__ in, h16* __restrict__ out,
                        int n) {
  int i = blockIdx.x * blockDim.x + threadIdx.x;
  if (i < n) out[i] = (h16)in[i];
}

__global__ void k_bias_sum(const float* a, const float* b, float* o, int n) {
  int i = blockIdx.x * blockDim.x + threadIdx.x;
  if (i < n) o[i] = a[i] + b[i];
}

// ---------------------------------------------------------------------------
// GEMM: C[M,N] = A[M,K](f16) * W[N,K](f16)^T. 64x64 block tile, 8 waves.
// mode 1: outH f16 row-major [M,N]          (Q, K projections)
// mode 2: outH f16 transposed [B,NH,64,S]   (V projection, for pass B frags)
// mode 3: outF f32 fragment-major [S,G,B]+bias (LSTM gate pre-activations)
// ---------------------------------------------------------------------------
__global__ __launch_bounds__(256) void k_gemm_f16(
    const h16* __restrict__ A, const h16* __restrict__ W,
    float* __restrict__ outF, h16* __restrict__ outH,
    const float* __restrict__ bias, int M, int N, int K, int mode) {
  __shared__ alignas(16) h16 As[64][40];   // [m][k], rows padded to 80B
  __shared__ alignas(16) h16 Bs[64][40];   // [n][k]
  int tid = threadIdx.x;
  int wave = tid >> 5, lane = tid & 31;
  int nblk = N >> 6;
  int m0 = (blockIdx.x / nblk) << 6;
  int n0 = (blockIdx.x % nblk) << 6;
  int id0 = wave * 2;
  int mt = id0 >> 2;
  int nt0 = id0 & 3;
  int lane15 = lane & 15;
  int ml = (mt << 4) + lane15;
  int abase = (lane & 16) ? 16 : 0;
  int bbase = (lane & 16) ? 16 : 0;
  int sr = tid >> 2;            // staging row 0..63
  int sc = (tid & 3) << 3;      // staging col 0,8,16,24
  v8f acc0 = {}, acc1 = {};

  for (int k0 = 0; k0 < K; k0 += 32) {
    __syncthreads();
    async_b128(&As[sr][sc], A + (size_t)(m0 + sr) * K + k0 + sc);
    async_b128(&Bs[sr][sc], W + (size_t)(n0 + sr) * K + k0 + sc);
    async_wait();
    __syncthreads();

    Frag16 af;
    af.u[0] = *(const uint4*)&As[ml][abase];
    af.u[1] = *(const uint4*)&As[ml][abase + 8];
#pragma unroll
    for (int t = 0; t < 2; ++t) {
      int nl = ((nt0 + t) << 4) + lane15;
      Frag16 bf;
      bf.u[0] = *(const uint4*)&Bs[nl][bbase];
      bf.u[1] = *(const uint4*)&Bs[nl][bbase + 8];
      if (t == 0) acc0 = wmma_f16(af.v, bf.v, acc0);
      else        acc1 = wmma_f16(af.v, bf.v, acc1);
    }
  }

#pragma unroll
  for (int t = 0; t < 2; ++t) {
    v8f acc = t ? acc1 : acc0;
    int col = n0 + ((nt0 + t) << 4) + lane15;
    int rowbase = m0 + (mt << 4) + ((lane & 16) ? 8 : 0);
    if (mode == 1) {
#pragma unroll
      for (int p = 0; p < 8; ++p)
        outH[(size_t)(rowbase + p) * N + col] = (h16)acc[p];
    } else if (mode == 2) {
      Frag8h tmp;
#pragma unroll
      for (int p = 0; p < 8; ++p) tmp.h[p] = (h16)acc[p];
      int b = rowbase >> 9;              // token / S
      int i = rowbase & (S_ - 1);
      size_t o = ((size_t)(b * NH_ + (col >> 6)) * 64 + (col & 63)) * S_ + i;
      *(uint4*)(outH + o) = tmp.u;
    } else {  // mode 3
      float bv = bias[col];
      int bbat = rowbase >> 9;
      int tbase = rowbase & (S_ - 1);
#pragma unroll
      for (int p = 0; p < 8; ++p)
        outF[((size_t)(tbase + p) * G_ + col) * B_ + bbat] = acc[p] + bv;
    }
  }
}

// ---------------------------------------------------------------------------
// LSTM recurrence. Block = (dir, batch-tile of 16); 512 sequential steps.
// gates[16,256] = pre(t) + h_prev @ Wh^T via WMMA; Wh [256n][64k] in LDS so
// B-fragments are contiguous; gate pre-acts arrive fragment-major [S,G,B].
// ---------------------------------------------------------------------------
__global__ __launch_bounds__(256) void k_lstm(
    const h16* __restrict__ wh_f, const h16* __restrict__ wh_b,
    const float* __restrict__ gpT_f, const float* __restrict__ gpT_b,
    h16* __restrict__ lstm_h) {
  __shared__ alignas(16) h16 wh[256][72];   // [n][k] (+8 pad) ~36.9 KB
  __shared__ float gbuf[16][256];           // 16 KB
  __shared__ alignas(16) h16 hbuf[16][72];  // [m][k] (+8 pad)
  __shared__ float cbuf[16][64];
  int tid = threadIdx.x, wave = tid >> 5, lane = tid & 31;
  int dir = blockIdx.x >> 2;
  int bt = blockIdx.x & 3;
  int lane15 = lane & 15;
  int abase = (lane & 16) ? 16 : 0;
  int bbase = (lane & 16) ? 16 : 0;
  int halfsel = (lane & 16) ? 8 : 0;
  const h16* whg = dir ? wh_b : wh_f;
  const float* gp = dir ? gpT_b : gpT_f;

  // stage Wh [256,64] f16 -> wh[256][72]
  for (int i = 0; i < 8; ++i) {
    int idx = i * 256 + tid;            // 2048 x 16B
    int rr = idx >> 3, seg = (idx & 7) << 3;
    *(uint4*)&wh[rr][seg] = *(const uint4*)(whg + rr * 64 + seg);
  }
  // zero h, c
#pragma unroll
  for (int e = 0; e < 4; ++e) {
    cbuf[(tid * 4 + e) >> 6][(tid * 4 + e) & 63] = 0.f;
    hbuf[(tid * 4 + e) >> 6][(tid * 4 + e) & 63] = (h16)0.f;
  }
  __syncthreads();

  for (int step = 0; step < S_; ++step) {
    int t = dir ? (S_ - 1 - step) : step;
    // A fragments: h_prev [16m x 64k] -> two 16x32 fragments, contiguous runs
    Frag16 a0, a1;
    a0.u[0] = *(const uint4*)&hbuf[lane15][abase];
    a0.u[1] = *(const uint4*)&hbuf[lane15][abase + 8];
    a1.u[0] = *(const uint4*)&hbuf[lane15][abase + 32];
    a1.u[1] = *(const uint4*)&hbuf[lane15][abase + 40];
#pragma unroll
    for (int u = 0; u < 2; ++u) {
      int n = ((wave * 2 + u) << 4) + lane15;
      // C init: gate pre-activations, fragment-major -> 2 x b128
      Frag8f cc;
      const float* src = gp + ((size_t)t * G_ + n) * B_ + bt * 16 + halfsel;
      cc.u[0] = *(const uint4*)src;
      cc.u[1] = *(const uint4*)(src + 4);
      v8f c = cc.f;
      Frag16 b0, b1;
      b0.u[0] = *(const uint4*)&wh[n][bbase];
      b0.u[1] = *(const uint4*)&wh[n][bbase + 8];
      b1.u[0] = *(const uint4*)&wh[n][bbase + 32];
      b1.u[1] = *(const uint4*)&wh[n][bbase + 40];
      c = wmma_f16(a0.v, b0.v, c);
      c = wmma_f16(a1.v, b1.v, c);
#pragma unroll
      for (int p = 0; p < 8; ++p) gbuf[p + halfsel][n] = c[p];
    }
    __syncthreads();
#pragma unroll
    for (int e = 0; e < 4; ++e) {
      int idx = tid + e * 256;
      int r = idx >> 6, u = idx & 63;
      float ig = gbuf[r][u], fg = gbuf[r][64 + u];
      float gg = gbuf[r][128 + u], og = gbuf[r][192 + u];
      float cn = sigf(fg) * cbuf[r][u] + sigf(ig) * tanhf(gg);
      float hn = sigf(og) * tanhf(cn);
      cbuf[r][u] = cn;
      hbuf[r][u] = (h16)hn;
      int b = bt * 16 + r;
      lstm_h[((size_t)b * S_ + t) * D2_ + dir * H_ + u] = (h16)hn;
    }
    __syncthreads();
  }
}

// ---------------------------------------------------------------------------
// Attention pass A: per-row softmax stats m_i, 1/Z_i (softmax over j).
// ---------------------------------------------------------------------------
__global__ __launch_bounds__(256) void k_attn_rowstats(
    const h16* __restrict__ kh, const h16* __restrict__ qh,
    const int* __restrict__ amask,
    float* __restrict__ mrow, float* __restrict__ invz) {
  __shared__ float ms[2][64];
  __shared__ float zs[2][64];
  int blk = blockIdx.x;
  int iblk = blk & 7;
  int hd = (blk >> 3) & 7;
  int b = blk >> 6;
  int tid = threadIdx.x, wave = tid >> 5, lane = tid & 31;
  int rt = wave >> 1;
  int par = wave & 1;
  int lane15 = lane & 15;

  int irow = iblk * 64 + rt * 16 + lane15;
  size_t krow = ((size_t)b * S_ + irow) * QN_ + hd * 64 + ((lane & 16) ? 16 : 0);
  Frag16 a0, a1;
  a0.u[0] = *(const uint4*)(kh + krow);
  a0.u[1] = *(const uint4*)(kh + krow + 8);
  a1.u[0] = *(const uint4*)(kh + krow + 32);
  a1.u[1] = *(const uint4*)(kh + krow + 40);

  float m[8], z[8];
#pragma unroll
  for (int p = 0; p < 8; ++p) { m[p] = -3.0e38f; z[p] = 0.f; }

  for (int jt = par; jt < 32; jt += 2) {
    int j = jt * 16 + lane15;
    size_t qrow = ((size_t)b * S_ + j) * QN_ + hd * 64 + ((lane & 16) ? 16 : 0);
    v16h b0 = *(const v16h*)(qh + qrow);
    v16h b1 = *(const v16h*)(qh + qrow + 32);
    v8f c = {};
    c = wmma_f16(a0.v, b0, c);
    c = wmma_f16(a1.v, b1, c);
    bool masked = (amask[b * S_ + j] == 0);
#pragma unroll
    for (int p = 0; p < 8; ++p) {
      float s = masked ? -1e9f : c[p] * 0.125f;
      float mn = fmaxf(m[p], s);
      z[p] = z[p] * __expf(m[p] - mn) + __expf(s - mn);
      m[p] = mn;
    }
  }
#pragma unroll
  for (int p = 0; p < 8; ++p) {
#pragma unroll
    for (int off = 1; off < 16; off <<= 1) {
      float mo = __shfl_xor(m[p], off, 32);
      float zo = __shfl_xor(z[p], off, 32);
      float mn = fmaxf(m[p], mo);
      z[p] = z[p] * __expf(m[p] - mn) + zo * __expf(mo - mn);
      m[p] = mn;
    }
  }
  if (lane15 == 0) {
#pragma unroll
    for (int p = 0; p < 8; ++p) {
      int r = rt * 16 + p + ((lane & 16) ? 8 : 0);
      ms[par][r] = m[p];
      zs[par][r] = z[p];
    }
  }
  __syncthreads();
  if (tid < 64) {
    float m0 = ms[0][tid], m1 = ms[1][tid];
    float mn = fmaxf(m0, m1);
    float zz = zs[0][tid] * __expf(m0 - mn) + zs[1][tid] * __expf(m1 - mn);
    size_t o = ((size_t)(b * NH_ + hd)) * S_ + iblk * 64 + tid;
    mrow[o] = mn;
    invz[o] = 1.f / zz;
  }
}

// ---------------------------------------------------------------------------
// Attention pass B: context = P^T V (WMMA chain), fused argmax + mean.
// V arrives pre-transposed [B,NH,64d,S] so its B-fragments are contiguous.
// ---------------------------------------------------------------------------
__global__ __launch_bounds__(256) void k_attn_ctx(
    const h16* __restrict__ kh, const h16* __restrict__ qh,
    const h16* __restrict__ vT, const int* __restrict__ amask,
    const float* __restrict__ mrow, const float* __restrict__ invz,
    float* __restrict__ ctxmean, float* __restrict__ importance) {
  __shared__ alignas(16) h16 Vt[64][40];   // [d][i] (+8 pad)
  __shared__ alignas(16) h16 Ps[64][40];   // [j][i] (+8 pad)
  __shared__ float mS[32];
  __shared__ float zS[32];
  int blk = blockIdx.x;
  int jblk = blk & 7;
  int hd = (blk >> 3) & 7;
  int b = blk >> 6;
  int tid = threadIdx.x, wave = tid >> 5, lane = tid & 31;
  int jt = wave >> 1;
  int dpair = wave & 1;
  int dt0 = dpair * 2;
  int lane15 = lane & 15;
  int jloc = jt * 16 + lane15;
  int jglob = jblk * 64 + jloc;
  bool producer = (dpair == 0);
  int abase = (lane & 16) ? 16 : 0;
  int bbase = (lane & 16) ? 16 : 0;
  int sr = tid >> 2;            // staging: d row
  int sc = (tid & 3) << 3;      // staging: i col
  const h16* vbase = vT + ((size_t)(b * NH_ + hd) * 64) * S_;

  v8f ctx0 = {}, ctx1 = {};
  size_t qrow = ((size_t)b * S_ + jglob) * QN_ + hd * 64 + ((lane & 16) ? 16 : 0);
  v16h q0 = *(const v16h*)(qh + qrow);
  v16h q1 = *(const v16h*)(qh + qrow + 32);
  bool masked = (amask[b * S_ + jglob] == 0);
  float bestv = -1.f;
  int besti = 0;

  for (int i0 = 0; i0 < S_; i0 += 32) {
    __syncthreads();
    async_b128(&Vt[sr][sc], vbase + (size_t)sr * S_ + i0 + sc);
    if (tid < 32) {
      size_t o = ((size_t)(b * NH_ + hd)) * S_ + i0 + tid;
      mS[tid] = mrow[o];
      zS[tid] = invz[o];
    }
    async_wait();
    __syncthreads();

    if (producer) {
#pragma unroll
      for (int it = 0; it < 2; ++it) {
        int irow = i0 + it * 16 + lane15;
        size_t krow = ((size_t)b * S_ + irow) * QN_ + hd * 64 + abase;
        Frag16 a0, a1;
        a0.u[0] = *(const uint4*)(kh + krow);
        a0.u[1] = *(const uint4*)(kh + krow + 8);
        a1.u[0] = *(const uint4*)(kh + krow + 32);
        a1.u[1] = *(const uint4*)(kh + krow + 40);
        v8f c = {};
        c = wmma_f16(a0.v, q0, c);
        c = wmma_f16(a1.v, q1, c);
#pragma unroll
        for (int p = 0; p < 8; ++p) {
          int il = it * 16 + p + ((lane & 16) ? 8 : 0);
          float s = masked ? -1e9f : c[p] * 0.125f;
          float pv = __expf(s - mS[il]) * zS[il];
          int ig = i0 + il;
          if (pv > bestv) { bestv = pv; besti = ig; }
          Ps[jloc][il] = (h16)pv;
        }
      }
    }
    __syncthreads();

    // A = P^T [16j x 32i] (contiguous runs), B = V [32i x 16d] from Vt rows
    Frag16 pa;
    pa.u[0] = *(const uint4*)&Ps[jloc][abase];
    pa.u[1] = *(const uint4*)&Ps[jloc][abase + 8];
#pragma unroll
    for (int u = 0; u < 2; ++u) {
      int dl = (dt0 + u) * 16 + lane15;
      Frag16 vb;
      vb.u[0] = *(const uint4*)&Vt[dl][bbase];
      vb.u[1] = *(const uint4*)&Vt[dl][bbase + 8];
      if (u == 0) ctx0 = wmma_f16(pa.v, vb.v, ctx0);
      else        ctx1 = wmma_f16(pa.v, vb.v, ctx1);
    }
  }

  float s0 = 0.f, s1 = 0.f;
#pragma unroll
  for (int p = 0; p < 8; ++p) { s0 += ctx0[p]; s1 += ctx1[p]; }
  s0 += __shfl_xor(s0, 16, 32);
  s1 += __shfl_xor(s1, 16, 32);
  if (lane < 16) {
    atomicAdd(&ctxmean[b * QN_ + hd * 64 + (dt0 + 0) * 16 + lane], s0);
    atomicAdd(&ctxmean[b * QN_ + hd * 64 + (dt0 + 1) * 16 + lane], s1);
  }
  if (producer) {
    float ov = __shfl_xor(bestv, 16, 32);
    int oi = __shfl_xor(besti, 16, 32);
    if (ov > bestv || (ov == bestv && oi < besti)) { bestv = ov; besti = oi; }
    if (lane < 16)
      importance[((size_t)(b * NH_ + hd)) * S_ + jglob] = (float)besti;
  }
}

// ---------------------------------------------------------------------------
// Final FC
// ---------------------------------------------------------------------------
__global__ void k_fc(const float* __restrict__ ctxmean,
                     const float* __restrict__ Wfc,
                     const float* __restrict__ bfc, float* __restrict__ out) {
  int b = blockIdx.x;
  int t = threadIdx.x;
  if (t >= TAR_) return;
  const float inv = 1.f / (float)S_;
  float acc = 0.f;
  for (int n = 0; n < QN_; ++n)
    acc += ctxmean[b * QN_ + n] * inv * Wfc[t * QN_ + n];
  out[b * TAR_ + t] = acc + bfc[t];
}

// ---------------------------------------------------------------------------
// Host-side orchestration
// ---------------------------------------------------------------------------
extern "C" void kernel_launch(void* const* d_in, const int* in_sizes, int n_in,
                              void* d_out, int out_size, void* d_ws,
                              size_t ws_size, hipStream_t stream) {
  (void)in_sizes; (void)n_in; (void)out_size; (void)ws_size;

  const int*   seq   = (const int*)d_in[0];
  const int*   amask = (const int*)d_in[1];
  const float* emb   = (const float*)d_in[2];
  const float* Wi_f  = (const float*)d_in[3];
  const float* Wh_f  = (const float*)d_in[4];
  const float* bi_f  = (const float*)d_in[5];
  const float* bh_f  = (const float*)d_in[6];
  const float* Wi_b  = (const float*)d_in[7];
  const float* Wh_b  = (const float*)d_in[8];
  const float* bi_b  = (const float*)d_in[9];
  const float* bh_b  = (const float*)d_in[10];
  const float* Wq    = (const float*)d_in[11];
  const float* Wk    = (const float*)d_in[12];
  const float* Wv    = (const float*)d_in[13];
  const float* Wfc   = (const float*)d_in[14];
  const float* bfc   = (const float*)d_in[15];

  float* out = (float*)d_out;
  float* imp = out + B_ * TAR_;

  size_t off = 0;
  auto carve = [&](size_t bytes) {
    size_t o = off;
    off = (off + bytes + 255) & ~(size_t)255;
    return o;
  };
  char* ws = (char*)d_ws;
  h16*   xh    = (h16*)(ws + carve((size_t)T_ * E_ * 2));
  h16*   wi16f = (h16*)(ws + carve((size_t)G_ * E_ * 2));
  h16*   wi16b = (h16*)(ws + carve((size_t)G_ * E_ * 2));
  h16*   wh16f = (h16*)(ws + carve((size_t)G_ * H_ * 2));
  h16*   wh16b = (h16*)(ws + carve((size_t)G_ * H_ * 2));
  h16*   wq16  = (h16*)(ws + carve((size_t)QN_ * D2_ * 2));
  h16*   wk16  = (h16*)(ws + carve((size_t)QN_ * D2_ * 2));
  h16*   wv16  = (h16*)(ws + carve((size_t)QN_ * D2_ * 2));
  float* bs_f  = (float*)(ws + carve((size_t)G_ * 4));
  float* bs_b  = (float*)(ws + carve((size_t)G_ * 4));
  float* gpT_f = (float*)(ws + carve((size_t)S_ * G_ * B_ * 4));
  float* gpT_b = (float*)(ws + carve((size_t)S_ * G_ * B_ * 4));
  h16*   lstmh = (h16*)(ws + carve((size_t)T_ * D2_ * 2));
  h16*   qh    = (h16*)(ws + carve((size_t)T_ * QN_ * 2));
  h16*   kh    = (h16*)(ws + carve((size_t)T_ * QN_ * 2));
  h16*   vT    = (h16*)(ws + carve((size_t)T_ * QN_ * 2));
  float* mrow  = (float*)(ws + carve((size_t)B_ * NH_ * S_ * 4));
  float* invz  = (float*)(ws + carve((size_t)B_ * NH_ * S_ * 4));
  float* ctxm  = (float*)(ws + carve((size_t)B_ * QN_ * 4));

  // weight / bias prep (layout-preserving f32 -> f16)
  k_tof16<<<(G_ * E_ + 255) / 256, 256, 0, stream>>>(Wi_f, wi16f, G_ * E_);
  k_tof16<<<(G_ * E_ + 255) / 256, 256, 0, stream>>>(Wi_b, wi16b, G_ * E_);
  k_tof16<<<(G_ * H_ + 255) / 256, 256, 0, stream>>>(Wh_f, wh16f, G_ * H_);
  k_tof16<<<(G_ * H_ + 255) / 256, 256, 0, stream>>>(Wh_b, wh16b, G_ * H_);
  k_tof16<<<(QN_ * D2_ + 255) / 256, 256, 0, stream>>>(Wq, wq16, QN_ * D2_);
  k_tof16<<<(QN_ * D2_ + 255) / 256, 256, 0, stream>>>(Wk, wk16, QN_ * D2_);
  k_tof16<<<(QN_ * D2_ + 255) / 256, 256, 0, stream>>>(Wv, wv16, QN_ * D2_);
  k_bias_sum<<<1, G_, 0, stream>>>(bi_f, bh_f, bs_f, G_);
  k_bias_sum<<<1, G_, 0, stream>>>(bi_b, bh_b, bs_b, G_);

  k_embed_gather<<<T_, E_, 0, stream>>>(seq, emb, xh);

  // LSTM input projections -> fragment-major [S,G,B] f32 (bias folded)
  k_gemm_f16<<<(T_ / 64) * (G_ / 64), 256, 0, stream>>>(
      xh, wi16f, gpT_f, nullptr, bs_f, T_, G_, E_, 3);
  k_gemm_f16<<<(T_ / 64) * (G_ / 64), 256, 0, stream>>>(
      xh, wi16b, gpT_b, nullptr, bs_b, T_, G_, E_, 3);

  // recurrence: 2 dirs x 4 batch tiles
  k_lstm<<<8, 256, 0, stream>>>(wh16f, wh16b, gpT_f, gpT_b, lstmh);

  // projections: Q,K row-major f16; V transposed [B,NH,64,S] f16
  k_gemm_f16<<<(T_ / 64) * (QN_ / 64), 256, 0, stream>>>(
      lstmh, wq16, nullptr, qh, nullptr, T_, QN_, D2_, 1);
  k_gemm_f16<<<(T_ / 64) * (QN_ / 64), 256, 0, stream>>>(
      lstmh, wk16, nullptr, kh, nullptr, T_, QN_, D2_, 1);
  k_gemm_f16<<<(T_ / 64) * (QN_ / 64), 256, 0, stream>>>(
      lstmh, wv16, nullptr, vT, nullptr, T_, QN_, D2_, 2);

  k_attn_rowstats<<<B_ * NH_ * (S_ / 64), 256, 0, stream>>>(kh, qh, amask,
                                                            mrow, invz);

  hipMemsetAsync(ctxm, 0, (size_t)B_ * QN_ * 4, stream);
  k_attn_ctx<<<B_ * NH_ * (S_ / 64), 256, 0, stream>>>(
      kh, qh, vT, amask, mrow, invz, ctxm, imp);

  k_fc<<<B_, 32, 0, stream>>>(ctxm, Wfc, bfc, out);
}
